// DSAM_conv_36034775613684
// MI455X (gfx1250) — matrix operand order repaired and checked by reading, not compiled, not run on hardware
//
#include <hip/hip_runtime.h>
#include <hip/hip_bf16.h>
#include <math.h>

typedef __attribute__((ext_vector_type(2)))  float     v2f;
typedef __attribute__((ext_vector_type(8)))  float     v8f;
typedef __attribute__((ext_vector_type(16))) _Float16  v16h;

#define Cc   256
#define CR   16
#define HH   256
#define WW   256
#define BB   4
#define PLANE (HH*WW)

// ---------------------------------------------------------------------------
// Kernel 1: strip pooling. One block per (b,c) plane (256x256 f32).
// Thread t owns column t (coalesced row reads); wave-level shuffle reduction
// produces per-wave row partials in LDS; column sums stay in registers.
// Outputs means: Sh[b,c,h] and Sv[b,c,w].
// ---------------------------------------------------------------------------
__global__ __launch_bounds__(256) void k1_strip_pool(
    const float* __restrict__ x, float* __restrict__ Sh, float* __restrict__ Sv) {
  __shared__ float rowpart[8 * HH];
  const int bc   = blockIdx.x;                // b*256 + c
  const float* plane = x + (size_t)bc * PLANE;
  const int tid  = threadIdx.x;
  const int wave = tid >> 5;
  const int lane = tid & 31;

  float colsum = 0.0f;
  for (int h = 0; h < HH; ++h) {
    float v = plane[h * WW + tid];
    colsum += v;
    float r = v;
    #pragma unroll
    for (int ofs = 16; ofs > 0; ofs >>= 1) r += __shfl_xor(r, ofs, 32);
    if (lane == 0) rowpart[wave * HH + h] = r;
  }
  Sv[(size_t)bc * WW + tid] = colsum * (1.0f / (float)HH);
  __syncthreads();
  float rs = 0.0f;
  #pragma unroll
  for (int w = 0; w < 8; ++w) rs += rowpart[w * HH + tid];
  Sh[(size_t)bc * HH + tid] = rs * (1.0f / (float)WW);
}

// ---------------------------------------------------------------------------
// Kernel 2: szh/szv = folded (BN o w1-conv) of the pooled strips, via
// v_wmma_f32_16x16x4_f32.  One wave per block.  grid = B * 16 tiles * 2 dirs.
//   A = w1 (M=16 out-ch x K=256 in-ch), B = S (K=256 x N=16 positions)
// A f32 16x4 layout: lane M=lane&15, VGPRs hold K=(half*2), (half*2)+1.
// B f32 4x16 layout mirrors A with N across lanes.
// D: VGPR r -> M = r + 8*(lane>=16), N = lane&15.
// ---------------------------------------------------------------------------
__global__ __launch_bounds__(32) void k2_strip_conv(
    const float* __restrict__ Sh, const float* __restrict__ Sv,
    const float* __restrict__ w1,
    const float* __restrict__ gamma, const float* __restrict__ beta,
    const float* __restrict__ mean,  const float* __restrict__ var,
    const float* __restrict__ wh,    const float* __restrict__ wv,
    float* __restrict__ szh, float* __restrict__ szv) {
  const int bid  = blockIdx.x;
  const int dir  = bid & 1;            // 0 = horizontal (Sh), 1 = vertical (Sv)
  const int rest = bid >> 1;
  const int b    = rest >> 4;
  const int pos0 = (rest & 15) << 4;
  const float* S = (dir ? Sv : Sh) + (size_t)b * Cc * 256;

  const int lane = threadIdx.x & 31;
  const int half = lane >> 4;
  const int mn   = lane & 15;

  v8f acc = {};
  for (int k0 = 0; k0 < Cc; k0 += 4) {
    const int k = k0 + half * 2;
    v2f a, bb;
    a.x  = w1[mn * Cc + k];
    a.y  = w1[mn * Cc + k + 1];
    bb.x = S[(size_t)k * 256 + pos0 + mn];
    bb.y = S[(size_t)(k + 1) * 256 + pos0 + mn];
    acc = __builtin_amdgcn_wmma_f32_16x16x4_f32(
        false, a, false, bb, (short)0, acc, false, false);
  }

  const float dscale = dir ? wv[0] : wh[0];
  float* dst = dir ? szv : szh;
  #pragma unroll
  for (int r = 0; r < 8; ++r) {
    const int o  = r + half * 8;
    const float s    = gamma[o] * rsqrtf(var[o] + 1e-5f);
    const float bias = dir ? 0.0f : (beta[o] - mean[o] * s);
    dst[((size_t)b * CR + o) * 256 + pos0 + mn] = dscale * s * acc[r] + bias;
  }
}

// ---------------------------------------------------------------------------
// Kernel 3: main fused kernel.  8 waves/block, one 16-pixel tile per wave.
// W2 (256x16) is staged once per block into LDS pre-swizzled into the
// v_wmma_f32_16x16x32_f16 A-operand layout (K = CR=16 zero-padded to 32):
//   lane<16 : elements 0..15 = K=0..15 (contiguous k order, matches B)
//   lane>=16: elements 0..15 = 0 (K=16..31 pad)
// B operand: lanes 0..15 hold K=0..15 = relu(szh[o]+szv[o]); lanes 16..31
// hold K=16..31 = all zero padding.
// Epilogue: exact GELU + out = x*(1 + ls*gelu).
// ---------------------------------------------------------------------------
__global__ __launch_bounds__(256) void k3_fused_main(
    const float* __restrict__ x,  const float* __restrict__ w2,
    const float* __restrict__ szh, const float* __restrict__ szv,
    const float* __restrict__ ls, float* __restrict__ out) {
  __shared__ __align__(32) _Float16 ldsA[16 * 32 * 16];  // 16 ctiles, 16 KB

  const int tid = threadIdx.x;
  // Stage W2 into WMMA A layout (each thread fills two 32-byte lane chunks).
  for (int idx = tid; idx < 512; idx += 256) {
    const int ct = idx >> 5;
    const int l  = idx & 31;
    const int m  = l & 15;
    const float* wrow = w2 + (size_t)(ct * 16 + m) * CR;   // w2[c][o]
    #pragma unroll
    for (int e = 0; e < 16; ++e) {
      float v = 0.0f;
      if (l < 16) v = wrow[e];
      ldsA[idx * 16 + e] = (_Float16)v;
    }
  }
  __syncthreads();

  const int wave = tid >> 5;
  const int lane = tid & 31;
  const int tile = blockIdx.x * 8 + wave;      // 16384 tiles total
  const int b    = tile >> 12;                 // 4096 tiles per batch
  const int rem  = tile & 4095;
  const int h    = rem >> 4;
  const int w0   = (rem & 15) << 4;
  const int col  = lane & 15;

  // Wave-uniform row strip values.
  float zh[CR];
  const float* szhp = szh + (size_t)b * CR * 256 + h;
  #pragma unroll
  for (int o = 0; o < CR; ++o) zh[o] = szhp[o * 256];

  // Build the f16 B operand (relu of strip sums); upper lane half = K padding.
  v16h bmat = {};
  if (lane < 16) {
    const float* szvp = szv + (size_t)b * CR * 256 + w0 + col;
    #pragma unroll
    for (int o = 0; o < CR; ++o) {
      float v = zh[o] + szvp[o * 256];
      bmat[o] = (_Float16)(v > 0.0f ? v : 0.0f);
    }
  }

  const size_t pixbase = (((size_t)b * Cc) * HH + h) * WW + w0 + col;
  const int choff = (lane >> 4) * 8;

  for (int ct = 0; ct < 16; ++ct) {
    v16h amat = *(const v16h*)(ldsA + (ct * 32 + lane) * 16);
    v8f acc = {};
    acc = __builtin_amdgcn_wmma_f32_16x16x32_f16(
        false, amat, false, bmat, (short)0, acc, false, false);
    #pragma unroll
    for (int r = 0; r < 8; ++r) {
      const int ch = ct * 16 + choff + r;         // D: M = r (+8 for hi lanes)
      const size_t addr = pixbase + (size_t)ch * PLANE;
      const float xv = x[addr];
      const float a  = acc[r];
      const float g  = 0.5f * a * (1.0f + erff(a * 0.70710678118654752f));
      out[addr] = xv + ls[ch] * xv * g;
    }
  }
}

// ---------------------------------------------------------------------------
extern "C" void kernel_launch(void* const* d_in, const int* in_sizes, int n_in,
                              void* d_out, int out_size, void* d_ws, size_t ws_size,
                              hipStream_t stream) {
  const float* x     = (const float*)d_in[0];
  const float* w1    = (const float*)d_in[1];
  const float* w2    = (const float*)d_in[2];
  const float* gamma = (const float*)d_in[3];
  const float* beta  = (const float*)d_in[4];
  const float* mean  = (const float*)d_in[5];
  const float* var   = (const float*)d_in[6];
  const float* wh    = (const float*)d_in[7];
  const float* wv    = (const float*)d_in[8];
  const float* ls    = (const float*)d_in[9];
  float* out = (float*)d_out;

  float* ws  = (float*)d_ws;
  float* Sh  = ws;                       // 4*256*256 = 262144 f32
  float* Sv  = ws + 262144;              // 262144 f32
  float* szh = ws + 524288;              // 4*16*256 = 16384 f32
  float* szv = ws + 540672;              // 16384 f32

  k1_strip_pool<<<BB * Cc, 256, 0, stream>>>(x, Sh, Sv);
  k2_strip_conv<<<BB * 16 * 2, 32, 0, stream>>>(Sh, Sv, w1, gamma, beta, mean,
                                                var, wh, wv, szh, szv);
  k3_fused_main<<<(BB * HH * WW / 16) / 8, 256, 0, stream>>>(x, w2, szh, szv,
                                                             ls, out);
}
